// MambaSequenceClassifierResidual_88459146428787
// MI455X (gfx1250) — compile-verified, matrix operand order checked by reading, not run on hardware
//
#include <hip/hip_runtime.h>

#define WS_M   32768      // B*L
#define SEQ_L  4096
#define NB     8
#define DSTATE 16
#define XSTR   544        // padded di stride (514 -> 544, mult of 32, 34 N-tiles)
#define LNSTR  288        // padded d stride  (257 -> 288, mult of 32)
#define XDSTR  64         // padded x_dbl stride (49 -> 64)

typedef __attribute__((ext_vector_type(16))) __bf16 v16bf;
typedef __attribute__((ext_vector_type(8)))  __bf16 v8bf;
typedef __attribute__((ext_vector_type(8)))  float  v8f;

__device__ __forceinline__ float sigm_f(float x) { return 1.0f / (1.0f + __expf(-x)); }
__device__ __forceinline__ float softplus_f(float x) { return (x > 20.f) ? x : log1pf(__expf(x)); }

// ---------------- embedding gather ----------------
__global__ void embed_kernel(const int* __restrict__ tok, const float* __restrict__ emb,
                             float* __restrict__ h) {
  int idx = blockIdx.x * blockDim.x + threadIdx.x;   // M*256 threads
  int m = idx >> 8, c = idx & 255;
  h[idx] = emb[(size_t)tok[m] * 256 + c];
}

// ---------------- layernorm, one wave32 per row ----------------
// MODE 0: bf16 out, zero-padded to dstride
// MODE 1: f32 out, d=256, dstride=257, col 256 = (float)token (residual concat)
// MODE 2: f32 out, cols < d
template<int MODE>
__global__ void ln_kernel(const float* __restrict__ src, int sstride, int d,
                          const float* __restrict__ w, const float* __restrict__ b,
                          void* __restrict__ dst, int dstride,
                          const int* __restrict__ tok) {
  int wid  = (blockIdx.x * blockDim.x + threadIdx.x) >> 5;
  int lane = threadIdx.x & 31;
  const float* row = src + (size_t)wid * sstride;
  float s = 0.f, s2 = 0.f;
  for (int i = lane; i < d; i += 32) { float v = row[i]; s += v; s2 += v * v; }
  for (int o = 16; o > 0; o >>= 1) { s += __shfl_xor(s, o, 32); s2 += __shfl_xor(s2, o, 32); }
  float inv  = 1.0f / (float)d;
  float mu   = s * inv;
  float rstd = rsqrtf(s2 * inv - mu * mu + 1e-5f);
  if (MODE == 0) {
    __bf16* o = (__bf16*)dst + (size_t)wid * dstride;
    for (int i = lane; i < dstride; i += 32) {
      float v = (i < d) ? (row[i] - mu) * rstd * w[i] + b[i] : 0.f;
      o[i] = (__bf16)v;
    }
  } else if (MODE == 1) {
    float* o = (float*)dst + (size_t)wid * dstride;
    for (int i = lane; i < dstride; i += 32)
      o[i] = (i < d) ? (row[i] - mu) * rstd * w[i] + b[i] : (float)tok[wid];
  } else {
    float* o = (float*)dst + (size_t)wid * dstride;
    for (int i = lane; i < d; i += 32)
      o[i] = (row[i] - mu) * rstd * w[i] + b[i];
  }
}

// ---------------- weight convert + zero-pad to bf16 ----------------
__global__ void convw_kernel(const float* __restrict__ W, int Nv, int K,
                             __bf16* __restrict__ dst, int Npad, int Kpad) {
  int idx = blockIdx.x * blockDim.x + threadIdx.x;
  if (idx >= Npad * Kpad) return;
  int n = idx / Kpad, k = idx % Kpad;
  float v = (n < Nv && k < K) ? W[(size_t)n * K + k] : 0.f;
  dst[idx] = (__bf16)v;
}

// ---------------- WMMA GEMM: C[M,N] = A[M,K] * W[N,K]^T ----------------
// one wave32 per 16x16 tile, 8 waves/block; bf16 inputs, f32 accum.
// EPI 0: store bf16, zero-fill cols >= Nvalid (keeps output usable as next A)
// EPI 1: accumulate f32 into Cout for cols < Nvalid (residual add)
template<int EPI>
__global__ void gemm_wmma(const __bf16* __restrict__ A, int lda,
                          const __bf16* __restrict__ W, int ldw,
                          void* __restrict__ Cout, int ldc,
                          int Ntiles, int Nvalid, int Kpad) {
  int wv   = (blockIdx.x * blockDim.x + threadIdx.x) >> 5;
  int lane = threadIdx.x & 31;
  int tN = wv % Ntiles, tM = wv / Ntiles;
  int l15 = lane & 15, half = lane >> 4, hi = half << 3;
  const __bf16* arow = A + (size_t)(tM * 16 + l15) * lda + hi;
  const __bf16* wrow = W + (size_t)(tN * 16 + l15) * ldw + hi;
  v8f acc = {0.f, 0.f, 0.f, 0.f, 0.f, 0.f, 0.f, 0.f};
  for (int k0 = 0; k0 < Kpad; k0 += 32) {
    v8bf a0 = *(const v8bf*)(arow + k0);
    v8bf a1 = *(const v8bf*)(arow + k0 + 16);
    v8bf b0 = *(const v8bf*)(wrow + k0);
    v8bf b1 = *(const v8bf*)(wrow + k0 + 16);
    v16bf av = __builtin_shufflevector(a0, a1, 0,1,2,3,4,5,6,7,8,9,10,11,12,13,14,15);
    v16bf bv = __builtin_shufflevector(b0, b1, 0,1,2,3,4,5,6,7,8,9,10,11,12,13,14,15);
    acc = __builtin_amdgcn_wmma_f32_16x16x32_bf16(false, av, false, bv,
                                                  (short)0, acc, false, false);
  }
  int nc = tN * 16 + l15;
#pragma unroll
  for (int r = 0; r < 8; ++r) {
    int mr = tM * 16 + r + hi;                 // D layout: VGPR r -> M = r + 8*half
    if (EPI == 0) {
      __bf16* C = (__bf16*)Cout;
      C[(size_t)mr * ldc + nc] = (__bf16)((nc < Nvalid) ? acc[r] : 0.f);
    } else {
      if (nc < Nvalid) {
        float* C = (float*)Cout;
        C[(size_t)mr * ldc + nc] += acc[r];
      }
    }
  }
}

// ---------------- depthwise causal conv1d (4 taps) + bias + SiLU ----------------
__global__ void conv_silu_kernel(const __bf16* __restrict__ xin,
                                 const float* __restrict__ cw, const float* __restrict__ cb,
                                 __bf16* __restrict__ xout, int di) {
  int idx = blockIdx.x * blockDim.x + threadIdx.x;   // M*XSTR threads
  int m = idx / XSTR, c = idx % XSTR;
  if (c >= di) { xout[idx] = (__bf16)0.f; return; }
  int t = m & (SEQ_L - 1);
  float acc = cb[c];
#pragma unroll
  for (int j = 0; j < 4; ++j) {
    int tt = t - 3 + j;
    float v = (tt >= 0) ? (float)xin[(size_t)(m - 3 + j) * XSTR + c] : 0.f;
    acc += v * cw[c * 4 + j];
  }
  acc = acc * sigm_f(acc);
  xout[idx] = (__bf16)acc;
}

// ---------------- negA = -exp(A_log) ----------------
__global__ void negA_kernel(const float* __restrict__ Alog, float* __restrict__ negA, int di) {
  int idx = blockIdx.x * blockDim.x + threadIdx.x;
  if (idx < di * DSTATE) negA[idx] = -__expf(Alog[idx]);
}

// ---------------- selective scan, one lane per (b, channel) ----------------
// folds: dt = softplus(dt_r . dt_w[d] + dt_b[d]); y += u*D; out = y*silu(z)
template<int R>
__global__ void scan_kernel(const __bf16* __restrict__ u, const __bf16* __restrict__ z,
                            const __bf16* __restrict__ xdb,
                            const float* __restrict__ dtw, const float* __restrict__ dtb,
                            const float* __restrict__ negA, const float* __restrict__ Dp,
                            __bf16* __restrict__ yg, int di) {
  int d = blockIdx.x * blockDim.x + threadIdx.x;
  int b = blockIdx.y;
  if (d >= di) return;
  float wdt[R];
#pragma unroll
  for (int j = 0; j < R; ++j) wdt[j] = dtw[d * R + j];
  float bias = dtb[d], Dv = Dp[d];
  float na[DSTATE], st[DSTATE];
#pragma unroll
  for (int s = 0; s < DSTATE; ++s) { na[s] = negA[d * DSTATE + s]; st[s] = 0.f; }
  size_t base = (size_t)b * SEQ_L;
  for (int t = 0; t < SEQ_L; ++t) {
    size_t m = base + t;
    const __bf16* xr = xdb + m * XDSTR;
    float dtv = bias;
#pragma unroll
    for (int j = 0; j < R; ++j) dtv += (float)xr[j] * wdt[j];
    dtv = softplus_f(dtv);
    float uv = (float)u[m * XSTR + d];
    float du = dtv * uv;
    float y = 0.f;
#pragma unroll
    for (int s = 0; s < DSTATE; ++s) {
      float Bc = (float)xr[R + s];
      float Cc = (float)xr[R + DSTATE + s];
      float dA = __expf(dtv * na[s]);
      st[s] = dA * st[s] + du * Bc;
      y += st[s] * Cc;
    }
    y += uv * Dv;
    float zv = (float)z[m * XSTR + d];
    yg[m * XSTR + d] = (__bf16)(y * (zv * sigm_f(zv)));
  }
}

// ---------------- mean pool over L ----------------
__global__ void pool_kernel(const float* __restrict__ src, float* __restrict__ pooled) {
  int idx = blockIdx.x * blockDim.x + threadIdx.x;
  if (idx >= NB * 257) return;
  int b = idx / 257, c = idx % 257;
  float s = 0.f;
  for (int t = 0; t < SEQ_L; ++t) s += src[((size_t)b * SEQ_L + t) * 257 + c];
  pooled[idx] = s * (1.0f / SEQ_L);
}

// ---------------- classifier (8x16, K=257) ----------------
__global__ void cls_kernel(const float* __restrict__ pooled, const float* __restrict__ cw,
                           const float* __restrict__ cbias, float* __restrict__ out) {
  int idx = threadIdx.x;            // 128 threads
  int b = idx >> 4, k = idx & 15;
  float s = cbias[k];
  for (int c = 0; c < 257; ++c) s += pooled[b * 257 + c] * cw[k * 257 + c];
  out[idx] = s;
}

// ---------------- one Mamba block on residual stream hres (f32) ----------------
static void run_mamba_block(hipStream_t stream, float* hres, int hstride, int d,
                            int di, int r,
                            const float* ln_w, const float* ln_b,
                            const float* in_w, const float* conv_w, const float* conv_b,
                            const float* xp_w, const float* dt_w, const float* dt_b,
                            const float* A_log, const float* Dp, const float* out_w,
                            __bf16* lnb, __bf16* xbuf, __bf16* zbuf, __bf16* xcv,
                            __bf16* xdb, float* negA,
                            __bf16* wbx, __bf16* wbz, __bf16* wxp, __bf16* wout) {
  const int M = WS_M;
  ln_kernel<0><<<M / 8, 256, 0, stream>>>(hres, hstride, d, ln_w, ln_b, lnb, LNSTR, nullptr);

  int nWx = XSTR * LNSTR;
  convw_kernel<<<(nWx + 255) / 256, 256, 0, stream>>>(in_w, di, d, wbx, XSTR, LNSTR);
  convw_kernel<<<(nWx + 255) / 256, 256, 0, stream>>>(in_w + (size_t)di * d, di, d, wbz, XSTR, LNSTR);
  int nWp = XDSTR * XSTR;
  convw_kernel<<<(nWp + 255) / 256, 256, 0, stream>>>(xp_w, r + 2 * DSTATE, di, wxp, XDSTR, XSTR);
  int nWo = LNSTR * XSTR;
  convw_kernel<<<(nWo + 255) / 256, 256, 0, stream>>>(out_w, d, di, wout, LNSTR, XSTR);

  int ntX = XSTR / 16;   // 34
  gemm_wmma<0><<<(M / 16) * ntX / 8, 256, 0, stream>>>(lnb, LNSTR, wbx, LNSTR, xbuf, XSTR, ntX, di, LNSTR);
  gemm_wmma<0><<<(M / 16) * ntX / 8, 256, 0, stream>>>(lnb, LNSTR, wbz, LNSTR, zbuf, XSTR, ntX, di, LNSTR);

  conv_silu_kernel<<<(M * XSTR) / 256, 256, 0, stream>>>(xbuf, conv_w, conv_b, xcv, di);

  int ntD = XDSTR / 16;  // 4
  gemm_wmma<0><<<(M / 16) * ntD / 8, 256, 0, stream>>>(xcv, XSTR, wxp, XSTR, xdb, XDSTR, ntD, r + 2 * DSTATE, XSTR);

  negA_kernel<<<(di * DSTATE + 255) / 256, 256, 0, stream>>>(A_log, negA, di);

  dim3 sg((di + 127) / 128, NB);
  if (r == 16)
    scan_kernel<16><<<sg, 128, 0, stream>>>(xcv, zbuf, xdb, dt_w, dt_b, negA, Dp, xbuf, di);
  else
    scan_kernel<17><<<sg, 128, 0, stream>>>(xcv, zbuf, xdb, dt_w, dt_b, negA, Dp, xbuf, di);

  int ntO = (d + 15) / 16;
  gemm_wmma<1><<<(M / 16) * ntO / 8, 256, 0, stream>>>(xbuf, XSTR, wout, XSTR, hres, hstride, ntO, d, XSTR);
}

extern "C" void kernel_launch(void* const* d_in, const int* in_sizes, int n_in,
                              void* d_out, int out_size, void* d_ws, size_t ws_size,
                              hipStream_t stream) {
  (void)in_sizes; (void)n_in; (void)out_size; (void)ws_size;
  const int*   tok    = (const int*)  d_in[0];
  const float* emb    = (const float*)d_in[1];
  const float* bln_w  = (const float*)d_in[2];
  const float* bln_b  = (const float*)d_in[3];
  const float* bin_w  = (const float*)d_in[4];
  const float* bcv_w  = (const float*)d_in[5];
  const float* bcv_b  = (const float*)d_in[6];
  const float* bxp_w  = (const float*)d_in[7];
  const float* bdt_w  = (const float*)d_in[8];
  const float* bdt_b  = (const float*)d_in[9];
  const float* bA     = (const float*)d_in[10];
  const float* bD     = (const float*)d_in[11];
  const float* bout_w = (const float*)d_in[12];
  const float* norm_w = (const float*)d_in[13];
  const float* norm_b = (const float*)d_in[14];
  const float* cln_w  = (const float*)d_in[15];
  const float* cln_b  = (const float*)d_in[16];
  const float* cin_w  = (const float*)d_in[17];
  const float* ccv_w  = (const float*)d_in[18];
  const float* ccv_b  = (const float*)d_in[19];
  const float* cxp_w  = (const float*)d_in[20];
  const float* cdt_w  = (const float*)d_in[21];
  const float* cdt_b  = (const float*)d_in[22];
  const float* cA     = (const float*)d_in[23];
  const float* cD     = (const float*)d_in[24];
  const float* cout_w = (const float*)d_in[25];
  const float* fin_w  = (const float*)d_in[26];
  const float* fin_b  = (const float*)d_in[27];
  const float* cls_w  = (const float*)d_in[28];
  const float* cls_b  = (const float*)d_in[29];

  uint8_t* p = (uint8_t*)d_ws;
  auto alloc = [&](size_t bytes) { void* q = p; p += (bytes + 255) & ~(size_t)255; return q; };
  float*  hbuf   = (float*) alloc((size_t)WS_M * 256 * 4);
  float*  hcat   = (float*) alloc((size_t)WS_M * 257 * 4);
  __bf16* lnb    = (__bf16*)alloc((size_t)WS_M * LNSTR * 2);
  __bf16* xbuf   = (__bf16*)alloc((size_t)WS_M * XSTR * 2);   // also gated-y
  __bf16* zbuf   = (__bf16*)alloc((size_t)WS_M * XSTR * 2);
  __bf16* xcv    = (__bf16*)alloc((size_t)WS_M * XSTR * 2);   // also f32 pool source
  __bf16* xdb    = (__bf16*)alloc((size_t)WS_M * XDSTR * 2);
  float*  negA   = (float*) alloc((size_t)XSTR * DSTATE * 4);
  float*  pooled = (float*) alloc((size_t)NB * 257 * 4);
  __bf16* wbx    = (__bf16*)alloc((size_t)XSTR * LNSTR * 2);
  __bf16* wbz    = (__bf16*)alloc((size_t)XSTR * LNSTR * 2);
  __bf16* wxp    = (__bf16*)alloc((size_t)XDSTR * XSTR * 2);
  __bf16* wout   = (__bf16*)alloc((size_t)LNSTR * XSTR * 2);

  embed_kernel<<<WS_M, 256, 0, stream>>>(tok, emb, hbuf);

  for (int i = 0; i < 4; ++i) {
    const int d = 256, di = 512, r = 16;
    run_mamba_block(stream, hbuf, 256, d, di, r,
                    bln_w + (size_t)i * d, bln_b + (size_t)i * d,
                    bin_w + (size_t)i * 2 * di * d,
                    bcv_w + (size_t)i * di * 4, bcv_b + (size_t)i * di,
                    bxp_w + (size_t)i * (r + 2 * DSTATE) * di,
                    bdt_w + (size_t)i * di * r, bdt_b + (size_t)i * di,
                    bA + (size_t)i * di * DSTATE, bD + (size_t)i * di,
                    bout_w + (size_t)i * d * di,
                    lnb, xbuf, zbuf, xcv, xdb, negA, wbx, wbz, wxp, wout);
  }

  // norm + concat residual (token as extra channel) -> hcat (stride 257)
  ln_kernel<1><<<WS_M / 8, 256, 0, stream>>>(hbuf, 256, 256, norm_w, norm_b, hcat, 257, tok);

  // combined block, d=257, di=514, r=17
  run_mamba_block(stream, hcat, 257, 257, 514, 17,
                  cln_w, cln_b, cin_w, ccv_w, ccv_b, cxp_w, cdt_w, cdt_b, cA, cD, cout_w,
                  lnb, xbuf, zbuf, xcv, xdb, negA, wbx, wbz, wxp, wout);

  // final LN -> f32 (reuse xcv region), mean pool, classifier
  float* poolsrc = (float*)xcv;
  ln_kernel<2><<<WS_M / 8, 256, 0, stream>>>(hcat, 257, 257, fin_w, fin_b, poolsrc, 257, nullptr);
  pool_kernel<<<(NB * 257 + 255) / 256, 256, 0, stream>>>(poolsrc, pooled);
  cls_kernel<<<1, 128, 0, stream>>>(pooled, cls_w, cls_b, (float*)d_out);
}
